// MultiheadAttention_601295422126
// MI455X (gfx1250) — matrix operand order repaired and checked by reading, not compiled, not run on hardware
//
#include <hip/hip_runtime.h>
#include <cmath>

// Problem constants (match reference)
#define BB 4
#define SS 2048
#define TT 2048
#define DD 1024
#define HH 8
#define EE 128

typedef __attribute__((ext_vector_type(16))) __bf16 v16bf;
typedef __attribute__((ext_vector_type(8)))  float  v8f;
typedef __attribute__((ext_vector_type(4)))  unsigned int u32x4;
typedef __attribute__((ext_vector_type(8)))  int i32x8;
typedef __attribute__((ext_vector_type(4)))  int i32x4;

struct alignas(16) U4 { unsigned int x, y, z, w; };
struct alignas(16) F4 { float x, y, z, w; };

union Frag {
  v16bf v;
  U4 q[2];
  unsigned short h[16];
};

// ---- CDNA5 feature detection (compile-time, with safe fallbacks) ----------
#if defined(__has_builtin)
#if __has_builtin(__builtin_amdgcn_global_load_async_to_lds_b128) && \
    __has_builtin(__builtin_amdgcn_s_wait_asynccnt)
#define USE_ASYNC_LDS 1
#endif
#if __has_builtin(__builtin_amdgcn_tensor_load_to_lds) && \
    __has_builtin(__builtin_amdgcn_s_wait_tensorcnt)
#define USE_TDM 1
#if __has_include(<hip/amd_detail/amd_gfx1250_TDM.h>)
#define TDM_6ARG 1
#endif
#endif
#endif

static __device__ __forceinline__ unsigned short f2bfu(float f) {
  __bf16 b = (__bf16)f;
  unsigned short u;
  __builtin_memcpy(&u, &b, 2);
  return u;
}

static __device__ __forceinline__ v8f vzero8() {
  v8f z;
#pragma unroll
  for (int i = 0; i < 8; ++i) z[i] = 0.0f;
  return z;
}

static __device__ __forceinline__ v8f wmma_bf16(const Frag& a, const Frag& b, v8f c) {
  // D = A(16x32 bf16) * B(32x16 bf16) + C(16x16 f32)
  return __builtin_amdgcn_wmma_f32_16x16x32_bf16(false, a.v, false, b.v,
                                                 (short)0, c, false, false);
}

#if defined(USE_ASYNC_LDS)
// Signature (from hipcc diagnostic): (v4i32 AS1* src, v4i32 AS3* dst, imm, imm)
static __device__ __forceinline__ void async_cp16(const unsigned short* g, unsigned short* l) {
  __builtin_amdgcn_global_load_async_to_lds_b128(
      (__attribute__((address_space(1))) i32x4*)g,
      (__attribute__((address_space(3))) i32x4*)l, 0, 0);
}
#endif

#if defined(USE_TDM)
// TDM: pull a 32(t) x 128(e) bf16 tile out of the [E=128, T=2048] tensor whose
// tile origin is gsrc (points at element [0, t0]); row stride = T elements.
// D# fields per CDNA5 ISA.
static __device__ __forceinline__ void tdm_load_v(const unsigned short* gsrc,
                                                  unsigned short* lds_dst) {
  unsigned long long ga = (unsigned long long)(__UINTPTR_TYPE__)gsrc;
  unsigned int lds_off =
      (unsigned int)(__UINTPTR_TYPE__)(__attribute__((address_space(3))) unsigned short*)lds_dst;
  u32x4 g0;
  g0[0] = 1u;                                      // count=1, user descriptor
  g0[1] = lds_off;                                 // lds_addr (bytes)
  g0[2] = (unsigned int)ga;                        // global_addr[31:0]
  g0[3] = (unsigned int)(ga >> 32) | (2u << 30);   // global_addr[56:32] | type=2
  i32x8 g1;
  g1[0] = 0x00010000;                 // workgroup_mask=0, data_size=1 (2 bytes)
  g1[1] = (int)(2048u << 16);         // tensor_dim0 = T = 2048 (bits 79:48, lo16)
  g1[2] = (int)(128u << 16);          // tensor_dim0 hi = 0 | tensor_dim1 = 128
  g1[3] = (int)(32u << 16);           // tensor_dim1 hi = 0 | tile_dim0 = 32
  g1[4] = 128;                        // tile_dim1 = 128, tile_dim2 = 0
  g1[5] = 2048;                       // tensor_dim0_stride = T (lo32)
  g1[6] = 0;                          // stride hi = 0, tensor_dim1_stride lo = 0
  g1[7] = 0;
  i32x4 z;
  z[0] = 0; z[1] = 0; z[2] = 0; z[3] = 0;
#if defined(TDM_6ARG)
  i32x8 z8;
#pragma unroll
  for (int i = 0; i < 8; ++i) z8[i] = 0;
  __builtin_amdgcn_tensor_load_to_lds(g0, g1, z, z, z8, 0);
#else
  __builtin_amdgcn_tensor_load_to_lds(g0, g1, z, z, 0);
#endif
}
#endif

// ---------------------------------------------------------------------------
// Kernel 0: elementwise f32 -> bf16 (vectorized x8). Removes all conversion
// VALU from the GEMM hot loops and 8x redundant f32 re-reads per head.
// ---------------------------------------------------------------------------
__global__ __launch_bounds__(256) void cvt_bf16_kernel(const float* __restrict__ src,
                                                       unsigned short* __restrict__ dst) {
  size_t i = (size_t)blockIdx.x * 256 + threadIdx.x;
  const F4* s = (const F4*)(src + i * 8);
  F4 a = s[0], b = s[1];
  alignas(16) unsigned short h[8] = {f2bfu(a.x), f2bfu(a.y), f2bfu(a.z), f2bfu(a.w),
                                     f2bfu(b.x), f2bfu(b.y), f2bfu(b.z), f2bfu(b.w)};
  *(U4*)(dst + i * 8) = *(const U4*)h;
}

// ---------------------------------------------------------------------------
// Kernel 1: pre-swizzle projection weights W[H][1024][128] (f32) into exact
// B-fragment order: out[((h*32+kc)*8+nt)*32+lane][16] bf16.
// ---------------------------------------------------------------------------
__global__ __launch_bounds__(256) void pack_w_kernel(const float* __restrict__ W,
                                                     unsigned short* __restrict__ out) {
  int idx  = blockIdx.x * 256 + threadIdx.x;   // 65536 total
  int lane = idx & 31;
  int nt   = (idx >> 5) & 7;
  int kc   = (idx >> 8) & 31;
  int h    = idx >> 13;
  int n     = nt * 16 + (lane & 15);
  int kbase = kc * 32 + 16 * (lane >> 4);
  const float* src = W + ((size_t)h * DD + kbase) * EE + n;
  alignas(16) unsigned short tmp[16];
#pragma unroll
  for (int j = 0; j < 16; ++j) tmp[j] = f2bfu(src[(size_t)j * EE]);
  U4* dst = (U4*)(out + (size_t)idx * 16);
  dst[0] = *(const U4*)&tmp[0];
  dst[1] = *(const U4*)&tmp[8];
}

// ---------------------------------------------------------------------------
// Kernel 2: projection GEMM. Out[b,h,s,e] = X[b,s,:] @ W[h,:,e] + bias[h,e].
// X is pre-converted bf16; W pre-packed bf16. All operands are b128 loads.
// TRANSPOSED=true stores [B,H,E,T] (for V) via contiguous b128 stores.
// ---------------------------------------------------------------------------
template <bool TRANSPOSED>
__global__ __launch_bounds__(256) void proj_kernel(const unsigned short* __restrict__ Xbf,
                                                   const unsigned short* __restrict__ Wpk,
                                                   const float* __restrict__ bias,
                                                   unsigned short* __restrict__ Out) {
  int bh = blockIdx.y;
  int b = bh >> 3, h = bh & 7;
  int wave = threadIdx.x >> 5, lane = threadIdx.x & 31;
  int lm = lane & 15, kh2 = lane >> 4;
  int m0 = blockIdx.x * 128 + wave * 16;

  const unsigned short* xrow = Xbf + ((size_t)b * SS + (m0 + lm)) * DD;
  const unsigned short* wbase = Wpk + (size_t)h * 131072;  // 32*8*32*16

  v8f acc[8];
#pragma unroll
  for (int i = 0; i < 8; ++i) acc[i] = vzero8();

  for (int kc = 0; kc < 32; ++kc) {
    int kk = kc * 32;
    Frag a;
    a.q[0] = *(const U4*)(xrow + kk + 16 * kh2);
    a.q[1] = *(const U4*)(xrow + kk + 16 * kh2 + 8);
    const unsigned short* wb = wbase + ((size_t)kc * 256 + lane) * 16;
#pragma unroll
    for (int nt = 0; nt < 8; ++nt) {
      Frag bf;
      bf.q[0] = *(const U4*)(wb + (size_t)nt * 512);
      bf.q[1] = *(const U4*)(wb + (size_t)nt * 512 + 8);
      acc[nt] = wmma_bf16(a, bf, acc[nt]);
    }
  }

  if (TRANSPOSED) {
#pragma unroll
    for (int nt = 0; nt < 8; ++nt) {
      int e = nt * 16 + lm;
      float bv = bias[h * EE + e];
      alignas(16) unsigned short hs[8];
#pragma unroll
      for (int r = 0; r < 8; ++r) hs[r] = f2bfu(acc[nt][r] + bv);
      U4* dst = (U4*)(Out + ((size_t)bh * EE + e) * TT + m0 + 8 * kh2);
      *dst = *(const U4*)hs;
    }
  } else {
#pragma unroll
    for (int nt = 0; nt < 8; ++nt) {
      int e = nt * 16 + lm;
      float bv = bias[h * EE + e];
#pragma unroll
      for (int r = 0; r < 8; ++r) {
        int row = m0 + r + 8 * kh2;
        Out[((size_t)bh * SS + row) * EE + e] = f2bfu(acc[nt][r] + bv);
      }
    }
  }
}

// ---------------------------------------------------------------------------
// Kernel 3: fused attention with double-buffered LDS staging of K/V tiles.
//   K tile: 32 rows x 128 e (contiguous 8KB slab of kh) -> async b128 copies.
//   V tile: 128 rows x 32 t (strided slice of vhT)      -> TDM descriptor.
// All 8 waves then consume the shared tiles via ds_load_b128; the score tile
// goes D -> LDS -> A-fragment for the second WMMA chain.
// ---------------------------------------------------------------------------
__global__ __launch_bounds__(256) void attn_kernel(const unsigned short* __restrict__ qh,
                                                   const unsigned short* __restrict__ kh,
                                                   const unsigned short* __restrict__ vhT,
                                                   unsigned short* __restrict__ merged) {
  __shared__ alignas(16) unsigned short lds_k[2][32 * 128];   // 2 x 8KB
  __shared__ alignas(16) unsigned short lds_v[2][128 * 32];   // 2 x 8KB
  __shared__ alignas(16) unsigned short lds_w[8 * 16 * 32];   // 8KB weight tiles

  int bh = blockIdx.y;
  int b = bh >> 3, h = bh & 7;
  int tid = threadIdx.x;
  int wave = tid >> 5, lane = tid & 31;
  int lm = lane & 15, kh2 = lane >> 4;
  int m0 = blockIdx.x * 128 + wave * 16;

  const unsigned short* qb = qh + (size_t)bh * SS * EE;
  const unsigned short* kb = kh + (size_t)bh * TT * EE;
  const unsigned short* vb = vhT + (size_t)bh * EE * TT;
  unsigned short* swv = lds_w + wave * 512;

  // Preload the wave's Q fragments (16 rows x full E), reused for all t
  Frag qf[4];
  const unsigned short* qrow = qb + (size_t)(m0 + lm) * EE;
#pragma unroll
  for (int c = 0; c < 4; ++c) {
    qf[c].q[0] = *(const U4*)(qrow + c * 32 + 16 * kh2);
    qf[c].q[1] = *(const U4*)(qrow + c * 32 + 16 * kh2 + 8);
  }

  v8f hacc[8];
#pragma unroll
  for (int i = 0; i < 8; ++i) hacc[i] = vzero8();

  const float scale = 0.0883883476483184f;  // 1/sqrt(128)

  // ---- tile staging helpers ----
  auto stage = [&](int t0, int buf) {
    // K tile: contiguous 8KB; 512 x b128, 2 per thread
#pragma unroll
    for (int r = 0; r < 2; ++r) {
      int i = tid + r * 256;
      const unsigned short* g = kb + (size_t)t0 * EE + i * 8;
      unsigned short* l = &lds_k[buf][i * 8];
#if defined(USE_ASYNC_LDS)
      async_cp16(g, l);
#else
      *(U4*)l = *(const U4*)g;
#endif
    }
    // V tile: 128 rows x 32 elements from [E,T] (row stride T)
#if defined(USE_TDM)
    if (wave == 0) tdm_load_v(vb + t0, &lds_v[buf][0]);
#else
#pragma unroll
    for (int r = 0; r < 2; ++r) {
      int i = tid + r * 256;
      int row = i >> 2, off = (i & 3) * 8;
      const unsigned short* g = vb + (size_t)row * TT + t0 + off;
      unsigned short* l = &lds_v[buf][row * 32 + off];
#if defined(USE_ASYNC_LDS)
      async_cp16(g, l);
#else
      *(U4*)l = *(const U4*)g;
#endif
    }
#endif
  };
  auto wait_staging = [&]() {
#if defined(USE_ASYNC_LDS)
    __builtin_amdgcn_s_wait_asynccnt(0);
#endif
#if defined(USE_TDM)
    __builtin_amdgcn_s_wait_tensorcnt(0);
#endif
  };

  stage(0, 0);
  wait_staging();
  __syncthreads();

  for (int t0 = 0; t0 < TT; t0 += 32) {
    int cur = (t0 >> 5) & 1;
    if (t0 + 32 < TT) stage(t0 + 32, cur ^ 1);

    const unsigned short* kt = &lds_k[cur][0];
    const unsigned short* vt = &lds_v[cur][0];

#pragma unroll
    for (int tt = 0; tt < 2; ++tt) {
      v8f sacc = vzero8();
      const unsigned short* krow = kt + (size_t)(tt * 16 + lm) * EE;
#pragma unroll
      for (int c = 0; c < 4; ++c) {
        Frag kf;
        kf.q[0] = *(const U4*)(krow + c * 32 + 16 * kh2);
        kf.q[1] = *(const U4*)(krow + c * 32 + 16 * kh2 + 8);
        sacc = wmma_bf16(qf[c], kf, sacc);
      }
      // weights = 1 + tanh(score/sqrt(E)); scatter into LDS [M][t_local]
#pragma unroll
      for (int r = 0; r < 8; ++r) {
        float w = 1.0f + tanhf(sacc[r] * scale);
        int M = r + 8 * kh2;
        swv[M * 32 + tt * 16 + lm] = f2bfu(w);
      }
    }
    // Reload the 16x32 weight tile as an A fragment
    Frag wf;
    wf.q[0] = *(const U4*)(swv + lm * 32 + 16 * kh2);
    wf.q[1] = *(const U4*)(swv + lm * 32 + 16 * kh2 + 8);
    // heads += W(16x32) @ V(32x128), V fragments from LDS
#pragma unroll
    for (int nt = 0; nt < 8; ++nt) {
      const unsigned short* vrow = vt + (size_t)(nt * 16 + lm) * 32 + 16 * kh2;
      Frag vf;
      vf.q[0] = *(const U4*)(vrow);
      vf.q[1] = *(const U4*)(vrow + 8);
      hacc[nt] = wmma_bf16(wf, vf, hacc[nt]);
    }

    wait_staging();
    __syncthreads();
  }

  // Store heads into merged [B,S,H*E] (bf16)
#pragma unroll
  for (int nt = 0; nt < 8; ++nt) {
    int col = h * EE + nt * 16 + lm;
#pragma unroll
    for (int r = 0; r < 8; ++r) {
      int row = m0 + r + 8 * kh2;
      merged[((size_t)b * SS + row) * DD + col] = f2bfu(hacc[nt][r]);
    }
  }
}

// ---------------------------------------------------------------------------
// Kernel 4: output projection out = merged @ Wo^T (f32 out).
// Rows of (pre-converted bf16) Wo are columns of Wo^T -> direct B fragments.
// ---------------------------------------------------------------------------
__global__ __launch_bounds__(256) void outproj_kernel(const unsigned short* __restrict__ merged,
                                                      const unsigned short* __restrict__ Wobf,
                                                      float* __restrict__ out) {
  int b = blockIdx.z;
  int c0 = blockIdx.y * 128;
  int wave = threadIdx.x >> 5, lane = threadIdx.x & 31;
  int lm = lane & 15, kh2 = lane >> 4;
  int m0 = blockIdx.x * 128 + wave * 16;

  const unsigned short* arow = merged + ((size_t)b * SS + m0 + lm) * DD;

  v8f acc[8];
#pragma unroll
  for (int i = 0; i < 8; ++i) acc[i] = vzero8();

  for (int kk = 0; kk < DD; kk += 32) {
    Frag a;
    a.q[0] = *(const U4*)(arow + kk + 16 * kh2);
    a.q[1] = *(const U4*)(arow + kk + 16 * kh2 + 8);
#pragma unroll
    for (int nt = 0; nt < 8; ++nt) {
      int i = c0 + nt * 16 + lm;
      const unsigned short* wrow = Wobf + (size_t)i * DD + kk + 16 * kh2;
      Frag bf;
      bf.q[0] = *(const U4*)(wrow);
      bf.q[1] = *(const U4*)(wrow + 8);
      acc[nt] = wmma_bf16(a, bf, acc[nt]);
    }
  }

#pragma unroll
  for (int nt = 0; nt < 8; ++nt) {
    int i = c0 + nt * 16 + lm;
#pragma unroll
    for (int r = 0; r < 8; ++r) {
      int row = m0 + r + 8 * kh2;
      out[((size_t)b * SS + row) * DD + i] = acc[nt][r];
    }
  }
}

// ---------------------------------------------------------------------------
// Launch. Workspace layout (bytes), total ~120 MB:
//   Wq_pk | Wk_pk | Wv_pk | Wo_bf (2MB each)
//   qbf | kbf | vbf | qh | kh | vhT | merged (16MB each)
// ---------------------------------------------------------------------------
extern "C" void kernel_launch(void* const* d_in, const int* in_sizes, int n_in,
                              void* d_out, int out_size, void* d_ws, size_t ws_size,
                              hipStream_t stream) {
  const float* q  = (const float*)d_in[0];
  const float* k  = (const float*)d_in[1];
  const float* v  = (const float*)d_in[2];
  const float* Wq = (const float*)d_in[3];
  const float* bq = (const float*)d_in[4];
  const float* Wk = (const float*)d_in[5];
  const float* bk = (const float*)d_in[6];
  const float* Wv = (const float*)d_in[7];
  const float* bv = (const float*)d_in[8];
  const float* Wo = (const float*)d_in[9];
  float* out = (float*)d_out;

  char* ws = (char*)d_ws;
  const size_t WPK = 2097152;    // H*1024*128*2 (== 1024*1024*2 for Wo)
  const size_t TEN = 16777216;   // B*S*D*2 == B*H*S*E*2
  unsigned short* Wq_pk  = (unsigned short*)(ws + 0 * WPK);
  unsigned short* Wk_pk  = (unsigned short*)(ws + 1 * WPK);
  unsigned short* Wv_pk  = (unsigned short*)(ws + 2 * WPK);
  unsigned short* Wo_bf  = (unsigned short*)(ws + 3 * WPK);
  unsigned short* qbf    = (unsigned short*)(ws + 4 * WPK);
  unsigned short* kbf    = (unsigned short*)(ws + 4 * WPK + 1 * TEN);
  unsigned short* vbf    = (unsigned short*)(ws + 4 * WPK + 2 * TEN);
  unsigned short* qh     = (unsigned short*)(ws + 4 * WPK + 3 * TEN);
  unsigned short* khb    = (unsigned short*)(ws + 4 * WPK + 4 * TEN);
  unsigned short* vhT    = (unsigned short*)(ws + 4 * WPK + 5 * TEN);
  unsigned short* merged = (unsigned short*)(ws + 4 * WPK + 6 * TEN);

  // f32 -> bf16 conversions (exact multiples of 2048 elements per block)
  cvt_bf16_kernel<<<4096, 256, 0, stream>>>(q, qbf);
  cvt_bf16_kernel<<<4096, 256, 0, stream>>>(k, kbf);
  cvt_bf16_kernel<<<4096, 256, 0, stream>>>(v, vbf);
  cvt_bf16_kernel<<<512, 256, 0, stream>>>(Wo, Wo_bf);

  pack_w_kernel<<<256, 256, 0, stream>>>(Wq, Wq_pk);
  pack_w_kernel<<<256, 256, 0, stream>>>(Wk, Wk_pk);
  pack_w_kernel<<<256, 256, 0, stream>>>(Wv, Wv_pk);

  dim3 pgrid(SS / 128, BB * HH);
  proj_kernel<false><<<pgrid, 256, 0, stream>>>(qbf, Wq_pk, bq, qh);
  proj_kernel<false><<<pgrid, 256, 0, stream>>>(kbf, Wk_pk, bk, khb);
  proj_kernel<true><<<pgrid, 256, 0, stream>>>(vbf, Wv_pk, bv, vhT);

  attn_kernel<<<dim3(SS / 128, BB * HH), 256, 0, stream>>>(qh, khb, vhT, merged);

  outproj_kernel<<<dim3(SS / 128, DD / 128, BB), 256, 0, stream>>>(merged, Wo_bf, out);
}
